// SGNO1d_51728586113325
// MI455X (gfx1250) — matrix operand order repaired and checked by reading, not compiled, not run on hardware
//
#include <hip/hip_runtime.h>

#define XP 16386   // padded spatial length (logical)
#define XPS 16392  // x row stride, multiple of 8 elems (16B)
#define XF 16384   // unpadded spatial length
#define BB 16      // batch
#define CW 64      // width (channels)
#define CH 128     // hidden width
#define NMODE 64
#define NBLK 4

typedef __attribute__((ext_vector_type(16))) __bf16 v16bf;
typedef __attribute__((ext_vector_type(8)))  float  v8f;
typedef __attribute__((ext_vector_type(4)))  unsigned int v4u;
typedef __attribute__((ext_vector_type(8)))  int v8i;
typedef __attribute__((ext_vector_type(4)))  int v4i;

#define LDA 40     // LDS tile row stride (80B = 64B row + 16B pad, 16B aligned)
#define MIDS 136   // Mid tile row stride (272B, 16B aligned)

#if __has_builtin(__builtin_amdgcn_tensor_load_to_lds)
#define USE_TDM 1
#else
#define USE_TDM 0
#endif

// ---------------------------------------------------------------- helpers ---
__device__ inline v8f zero8() {
  v8f z = {0.f, 0.f, 0.f, 0.f, 0.f, 0.f, 0.f, 0.f};
  return z;
}
__device__ inline float geluf(float t) {
  return 0.5f * t * (1.f + erff(t * 0.70710678118654752f));
}
__device__ inline float softplusf(float x) {
  return x > 20.f ? x : log1pf(expf(x));
}
__device__ inline float phi1f(float z) {
  float az = fabsf(z);
  if (az < 1e-6f) return 1.f + 0.5f * z + z * z * (1.f / 6.f);
  return (expf(z) - 1.f) / z;
}

__device__ inline v8f wmma_bf16(v16bf a, v16bf b, v8f c) {
  return __builtin_amdgcn_wmma_f32_16x16x32_bf16(
      false, a, false, b, (short)0, c, false, false);
}

union Pk8 { uint4 q; __bf16 h8[8]; };
union Frag { v16bf v; uint4 q[2]; };

// A fragment (16x32 bf16): per ISA layout each lane's K-runs are contiguous.
__device__ inline v16bf load_afrag(const __bf16* As, int mt, int lane) {
  const __bf16* row = As + (mt * 16 + (lane & 15)) * LDA;
  int kh = (lane >> 4) << 3;
  Frag u;
  u.q[0] = *(const uint4*)(row + kh);
  u.q[1] = *(const uint4*)(row + 16 + kh);
  return u.v;
}
// B fragment (32x16 bf16) from fragment-major tile (row = n, contiguous K).
__device__ inline v16bf load_bfrag(const __bf16* Bs, int stride, int nt, int lane) {
  const __bf16* row = Bs + (nt * 16 + (lane & 15)) * stride + ((lane >> 4) << 3);
  Frag u;
  u.q[0] = *(const uint4*)(row);
  u.q[1] = *(const uint4*)(row + 16);
  return u.v;
}

// Fallback VGPR-mediated staging (rows x 32, zero-fill OOB).
__device__ inline void stageT(__bf16* dst, const __bf16* src, long ld,
                              int rows, int rval, int kval, int tid) {
  int tot = rows * 4;
  for (int idx = tid; idx < tot; idx += 256) {
    int m = idx >> 2, c = (idx & 3) << 3;
    __bf16* d = dst + m * LDA + c;
    if (m < rval) {
      const __bf16* s = src + (long)m * ld + c;
      if (c + 8 <= kval) {
        *(uint4*)d = *(const uint4*)s;
      } else {
        for (int j = 0; j < 8; ++j) d[j] = (c + j < kval) ? s[j] : (__bf16)0.f;
      }
    } else {
      *(uint4*)d = make_uint4(0u, 0u, 0u, 0u);
    }
  }
}

#if USE_TDM
// TDM 2D tile load: tile (tile1 rows x 32 bf16) from row-major source with row
// stride `ld` elems; valid extent (rval rows x kval elems) -> OOB returns zero.
// LDS dest rows padded 64B->80B via pad_interval=3 (16 dw), pad_amount=3 (4 dw).
__device__ inline void tdm2d(unsigned lds_addr, const void* gaddr,
                             unsigned kval, unsigned rval,
                             unsigned tile1, unsigned ld) {
  unsigned long ga = (unsigned long)gaddr;
  v4u g0;
  g0[0] = 1u;                                   // count=1, user descriptor
  g0[1] = lds_addr;                             // lds_addr [63:32]
  g0[2] = (unsigned)ga;                         // global_addr lo
  g0[3] = (unsigned)((ga >> 32) & 0x01FFFFFFu)  // global_addr hi (57-bit)
          | 0x80000000u;                        // type=2 ("image") [127:126]
  v8i g1;
  g1[0] = (int)((1u << 16)      // data_size = 1 -> 2-byte elements
                | (1u << 20)    // pad_enable
                | (3u << 22)    // pad_interval: 16 DWORDs (64B)
                | (3u << 25));  // pad_amount: 4 DWORDs (16B)
  g1[1] = (int)((kval & 0xFFFFu) << 16);                 // tensor_dim0 lo16
  g1[2] = (int)((kval >> 16) | ((rval & 0xFFFFu) << 16)); // dim0 hi / dim1 lo
  g1[3] = (int)((rval >> 16) | (32u << 16));             // dim1 hi / tile_dim0=32
  g1[4] = (int)(tile1 & 0xFFFFu);                        // tile_dim1 (tile_dim2=0)
  g1[5] = (int)ld;                                       // tensor_dim0_stride lo
  g1[6] = 0;
  g1[7] = 0;
  v4i gz = {0, 0, 0, 0};
  v8i gz8 = {0, 0, 0, 0, 0, 0, 0, 0};
  __builtin_amdgcn_tensor_load_to_lds(g0, g1, gz, gz, gz8, 0);
}
#endif

// Issue tile load (TDM by wave 0, or fallback all-thread staging).
__device__ inline void tile_load(__bf16* dst, const __bf16* src, long ld,
                                 int rows, int rval, int kval, int tid) {
#if USE_TDM
  if (tid < 32)
    tdm2d((unsigned)(unsigned long)(void*)dst, (const void*)src,
          (unsigned)kval, (unsigned)rval, (unsigned)rows, (unsigned)ld);
#else
  stageT(dst, src, ld, rows, rval, kval, tid);
#endif
}
// Wait for all outstanding TDM tile loads (before the workgroup barrier).
__device__ inline void tile_wait(int tid) {
#if USE_TDM
  if (tid < 32) __builtin_amdgcn_s_wait_tensorcnt((short)0);
#endif
}

// Transposing A-stage for the forward DFT: A[c][k] (64x32) from x-major act.
__device__ inline void stageAT(__bf16* As, const __bf16* hx, int kval, int tid) {
  int k = tid >> 3, c0 = (tid & 7) << 3;  // 32 k x 8 chunks = 256 threads
  if (k < kval) {
    Pk8 u;
    u.q = *(const uint4*)(hx + (long)k * 64 + c0);
#pragma unroll
    for (int j = 0; j < 8; ++j) As[(c0 + j) * LDA + k] = u.h8[j];
  } else {
#pragma unroll
    for (int j = 0; j < 8; ++j) As[(c0 + j) * LDA + k] = (__bf16)0.f;
  }
}

// ---------------------------------------------------------------- prep ------
__global__ void k_prep_scalars(const float* awr, const float* agr, float* scal) {
  if (threadIdx.x == 0) {
    scal[0] = softplusf(awr[0]);  // aw
    scal[1] = softplusf(agr[0]);  // ag
  }
}

__global__ void k_dft_tables(__bf16* DFt, __bf16* DIt) {
  int x = blockIdx.x * 256 + threadIdx.x;
  if (x >= XP) return;
  const float w0 = 6.28318530717958647692f / (float)XP;
  for (int m = 0; m < NMODE; ++m) {
    int ph = (m * x) % XP;
    float th = w0 * (float)ph;
    float s, c;
    __sincosf(th, &s, &c);
    DFt[(long)m * XPS + x] = (__bf16)c;
    DFt[(long)(64 + m) * XPS + x] = (__bf16)(-s);
    float cm = (m == 0 ? 1.f : 2.f) / (float)XP;
    DIt[(long)x * 128 + m] = (__bf16)(cm * c);
    DIt[(long)x * 128 + 64 + m] = (__bf16)(-cm * s);
  }
}

__global__ void k_etd_tables(const float* log_decay, float* expz, float* pf) {
  int id = blockIdx.x * 256 + threadIdx.x;
  if (id >= NBLK * CW * NMODE) return;
  int m = id & 63;
  float z = -softplusf(log_decay[id]);  // DT = 1.0
  expz[id] = expf(z);
  float gm = (float)m / 63.f;
  float r = sqrtf(gm * gm + 1e-12f);
  float r2 = r * r, r4 = r2 * r2, r8 = r4 * r4;
  pf[id] = phi1f(z) * expf(-2.0f * r8);
}

__global__ void k_weights(const float* g_w1, const float* g_w2, const float* w_w,
                          const float* fc1_w, const float* w_b, const float* g_b2,
                          const float* scal, __bf16* W1B, __bf16* W2B, __bf16* WWB,
                          __bf16* FC1B, float* WBS, float* B2S) {
  int id = blockIdx.x * 256 + threadIdx.x;
  float aw = scal[0], ag = scal[1];
  if (id < 32768) { W1B[id] = (__bf16)g_w1[id]; return; }
  id -= 32768;
  if (id < 32768) { W2B[id] = (__bf16)(ag * g_w2[id]); return; }
  id -= 32768;
  if (id < 16384) { WWB[id] = (__bf16)(aw * w_w[id]); return; }
  id -= 16384;
  if (id < 8192) { FC1B[id] = (__bf16)fc1_w[id]; return; }
  id -= 8192;
  if (id < 256) { WBS[id] = aw * w_b[id]; return; }
  id -= 256;
  if (id < 256) { B2S[id] = ag * g_b2[id]; return; }
}

__global__ void k_mix(const float* mre, const float* mim, __bf16* MXRE,
                      __bf16* MXIM, __bf16* MXIMN) {
  int id = blockIdx.x * 256 + threadIdx.x;
  if (id >= NBLK * 64 * 64 * 64) return;
  int ii = id & 63, o = (id >> 6) & 63, m = (id >> 12) & 63, i = id >> 18;
  long s = (((long)(i * 64 + ii) * 64 + o) * 64 + m);
  MXRE[id] = (__bf16)mre[s];
  float v = mim[s];
  MXIM[id] = (__bf16)v;
  MXIMN[id] = (__bf16)(-v);
}

// ------------------------------------------------------- fc0 (lift) ---------
__global__ void k_fc0(const float* x, const float* grid, const float* w,
                      const float* bia, __bf16* hX) {
  long id = (long)blockIdx.x * 256 + threadIdx.x;
  if (id >= (long)BB * XP) return;
  int b = (int)(id / XP), xx = (int)(id % XP);
  __bf16* hp = hX + ((long)b * XP + xx) * 64;
  if (xx >= XF) {
    for (int c = 0; c < 8; ++c) *(uint4*)(hp + c * 8) = make_uint4(0u, 0u, 0u, 0u);
    return;
  }
  float fe[31];
  const float* xin = x + ((long)b * XF + xx) * 30;
#pragma unroll
  for (int j = 0; j < 30; ++j) fe[j] = xin[j];
  fe[30] = grid[(long)b * XF + xx];
  for (int c = 0; c < 8; ++c) {
    Pk8 u;
    for (int r = 0; r < 8; ++r) {
      int o = c * 8 + r;
      float a = bia[o];
      const float* wr = w + o * 31;
#pragma unroll
      for (int j = 0; j < 31; ++j) a += wr[j] * fe[j];
      u.h8[r] = (__bf16)a;
    }
    *(uint4*)(hp + c * 8) = u.q;
  }
}

// --------------------------------------- fused g = ag*(W2 gelu(W1 h + b1) + b2)
__global__ __launch_bounds__(256) void k_gfused(const __bf16* hX, __bf16* gX,
    const __bf16* W1B, const float* gb1, const __bf16* W2B, const float* B2S,
    int blk) {
  __shared__ __bf16 As[128 * LDA];
  __shared__ __bf16 Bs[64 * LDA];
  __shared__ __bf16 Mid[64 * MIDS];  // fragment-major: [n][o]
  int n0 = blockIdx.x * 64, b = blockIdx.y;
  int tid = threadIdx.x, w = tid >> 5, lane = tid & 31;
  int ncols = XP - n0; if (ncols > 64) ncols = 64;
  const __bf16* W1 = W1B + blk * CH * CW;
  const __bf16* hxb = hX + ((long)b * XP + n0) * 64;
  v8f acc[4];
  for (int j = 0; j < 4; ++j) acc[j] = zero8();
  for (int kk = 0; kk < CW; kk += 32) {
    tile_load(As, W1 + kk, CW, 128, 128, 32, tid);
    tile_load(Bs, hxb + kk, 64, 64, ncols, 32, tid);
    tile_wait(tid);
    __syncthreads();
    v16bf af = load_afrag(As, w, lane);
#pragma unroll
    for (int nt = 0; nt < 4; ++nt) {
      v16bf bf = load_bfrag(Bs, LDA, nt, lane);
      acc[nt] = wmma_bf16(af, bf, acc[nt]);
    }
    __syncthreads();
  }
  int o0 = w * 16 + 8 * (lane >> 4);
  for (int nt = 0; nt < 4; ++nt) {
    int n = nt * 16 + (lane & 15);
    Pk8 u;
#pragma unroll
    for (int r = 0; r < 8; ++r)
      u.h8[r] = (__bf16)geluf(acc[nt][r] + gb1[blk * CH + o0 + r]);
    *(uint4*)(Mid + n * MIDS + o0) = u.q;
  }
  __syncthreads();
  const __bf16* W2 = W2B + blk * CW * CH;
  v8f a2[2]; a2[0] = zero8(); a2[1] = zero8();
  int mt = w >> 1, ntb = (w & 1) * 2;
  for (int kk = 0; kk < CH; kk += 32) {
    tile_load(As, W2 + kk, CH, 64, 64, 32, tid);
    tile_wait(tid);
    __syncthreads();
    v16bf af = load_afrag(As, mt, lane);
#pragma unroll
    for (int s = 0; s < 2; ++s) {
      v16bf bf = load_bfrag(Mid + kk, MIDS, ntb + s, lane);
      a2[s] = wmma_bf16(af, bf, a2[s]);
    }
    __syncthreads();
  }
  __bf16* gxb = gX + (long)b * XP * 64;
  int p0 = mt * 16 + 8 * (lane >> 4);
  for (int s = 0; s < 2; ++s) {
    int n = n0 + (ntb + s) * 16 + (lane & 15);
    if (n < XP) {
      Pk8 u;
#pragma unroll
      for (int r = 0; r < 8; ++r)
        u.h8[r] = (__bf16)(a2[s][r] + B2S[blk * CW + p0 + r]);
      *(uint4*)(gxb + (long)n * 64 + p0) = u.q;
    }
  }
}

// ------------------------------------------------ forward DFT (split-K, atomic)
__global__ __launch_bounds__(256) void k_dft(const __bf16* actX, const __bf16* DFt,
                                             float* C) {
  __shared__ __bf16 As[64 * LDA];
  __shared__ __bf16 Bs[64 * LDA];
  int bx = blockIdx.x;              // batch (A row block = 64 channels)
  int n0 = blockIdx.y * 64;         // mode-col block (0 or 64)
  int kb = blockIdx.z * 2080;       // split-K chunk
  int tid = threadIdx.x, w = tid >> 5, lane = tid & 31;
  int mt = w >> 1, ntb = (w & 1) * 2;
  const __bf16* Ax = actX + (long)bx * XP * 64;
  v8f acc[2]; acc[0] = zero8(); acc[1] = zero8();
  for (int s = 0; s < 65; ++s) {
    int kk = kb + s * 32;
    if (kk >= XP) break;
    int kv = XP - kk; if (kv > 32) kv = 32;
    tile_load(Bs, DFt + (long)n0 * XPS + kk, XPS, 64, 64, kv, tid);
    stageAT(As, Ax + (long)kk * 64, kv, tid);  // overlaps with TDM transfer
    int kn = kk + 32;
    if (kn < XP && tid < 32)
      __builtin_prefetch(Ax + (long)(kn + tid) * 64, 0, 1);
    tile_wait(tid);
    __syncthreads();
    v16bf af = load_afrag(As, mt, lane);
#pragma unroll
    for (int s2 = 0; s2 < 2; ++s2) {
      v16bf bf = load_bfrag(Bs, LDA, ntb + s2, lane);
      acc[s2] = wmma_bf16(af, bf, acc[s2]);
    }
    __syncthreads();
  }
  for (int s2 = 0; s2 < 2; ++s2)
    for (int r = 0; r < 8; ++r) {
      int row = bx * 64 + mt * 16 + r + 8 * (lane >> 4);
      int col = n0 + (ntb + s2) * 16 + (lane & 15);
      atomicAdd(&C[(long)row * 128 + col], acc[s2][r]);
    }
}

__global__ void k_zero(float* a, int n) {
  int i = blockIdx.x * 256 + threadIdx.x;
  if (i < n) a[i] = 0.f;
}

__global__ void k_gsplit(const float* GH, __bf16* GRE, __bf16* GIM) {
  int id = blockIdx.x * 256 + threadIdx.x;
  if (id >= 64 * 16 * 64) return;
  int i = id & 63, b = (id >> 6) & 15, m = id >> 10;
  long s = (long)(b * 64 + i) * 128;
  GRE[id] = (__bf16)GH[s + m];
  GIM[id] = (__bf16)GH[s + 64 + m];
}

// per-mode complex 16x64x64: re = Gre@Mre + Gim@(-Mim); im = Gre@Mim + Gim@Mre
__global__ __launch_bounds__(256) void k_modemix(const __bf16* GRE, const __bf16* GIM,
    const __bf16* MXRE, const __bf16* MXIM, const __bf16* MXIMN,
    float* GMR, float* GMI, int blk) {
  __shared__ __bf16 Ar[16 * LDA];
  __shared__ __bf16 Ai[16 * LDA];
  __shared__ __bf16 Br[64 * LDA];
  __shared__ __bf16 Bi[64 * LDA];
  __shared__ __bf16 Bn[64 * LDA];
  int m = blockIdx.x;
  int tid = threadIdx.x, w = tid >> 5, lane = tid & 31;
  int isIm = w >> 2, nt = w & 3;
  const __bf16* gre = GRE + m * 16 * 64;
  const __bf16* gim = GIM + m * 16 * 64;
  long mb = ((long)blk * 64 + m) * 64 * 64;
  v8f acc = zero8();
  for (int kk = 0; kk < 64; kk += 32) {
    tile_load(Ar, gre + kk, 64, 16, 16, 32, tid);
    tile_load(Ai, gim + kk, 64, 16, 16, 32, tid);
    tile_load(Br, MXRE + mb + kk, 64, 64, 64, 32, tid);
    tile_load(Bi, MXIM + mb + kk, 64, 64, 64, 32, tid);
    tile_load(Bn, MXIMN + mb + kk, 64, 64, 64, 32, tid);
    tile_wait(tid);
    __syncthreads();
    v16bf aR = load_afrag(Ar, 0, lane);
    v16bf aI = load_afrag(Ai, 0, lane);
    v16bf b1 = load_bfrag(isIm ? Bi : Br, LDA, nt, lane);
    v16bf b2 = load_bfrag(isIm ? Br : Bn, LDA, nt, lane);
    acc = wmma_bf16(aR, b1, acc);
    acc = wmma_bf16(aI, b2, acc);
    __syncthreads();
  }
  float* dst = isIm ? GMI : GMR;
  for (int r = 0; r < 8; ++r) {
    int b = r + 8 * (lane >> 4);
    int o = nt * 16 + (lane & 15);
    dst[(long)(m * 16 + b) * 64 + o] = acc[r];
  }
}

__global__ void k_combine(const float* VH, const float* GMR, const float* GMI,
                          const float* EXPZ, const float* PF, __bf16* OUTM,
                          int blk) {
  int id = blockIdx.x * 256 + threadIdx.x;
  if (id >= 16 * 64 * 64) return;
  int m = id & 63, o = (id >> 6) & 63, b = id >> 12;
  float ez = EXPZ[((blk * 64 + o) << 6) + m];
  float pv = PF[((blk * 64 + o) << 6) + m];
  long vr = (long)(b * 64 + o) * 128;
  long gi = (long)(m * 16 + b) * 64 + o;
  OUTM[vr + m] = (__bf16)(ez * VH[vr + m] + pv * GMR[gi]);
  OUTM[vr + 64 + m] = (__bf16)(ez * VH[vr + 64 + m] + pv * GMI[gi]);
}

// fused update: h' = act( OutM@Dinv + aw*(Ww@h) + aw*wb ), unified K=192 loop
__global__ __launch_bounds__(256) void k_update(const __bf16* OUTM, const __bf16* DIt,
    const __bf16* WWB, const __bf16* hX, const float* WBS, __bf16* hXn,
    int blk, int last) {
  __shared__ __bf16 As[64 * LDA];
  __shared__ __bf16 Bs[64 * LDA];
  int n0 = blockIdx.x * 64, b = blockIdx.y;
  int tid = threadIdx.x, w = tid >> 5, lane = tid & 31;
  int ncols = XP - n0; if (ncols > 64) ncols = 64;
  int mt = w >> 1, ntb = (w & 1) * 2;
  const __bf16* Ao = OUTM + (long)b * 64 * 128;
  const __bf16* Ww = WWB + blk * CW * CW;
  const __bf16* hxb = hX + ((long)b * XP + n0) * 64;
  v8f acc[2]; acc[0] = zero8(); acc[1] = zero8();
  for (int step = 0; step < 6; ++step) {
    if (step < 4) {
      int kk = step * 32;
      tile_load(As, Ao + kk, 128, 64, 64, 32, tid);
      tile_load(Bs, DIt + (long)n0 * 128 + kk, 128, 64, ncols, 32, tid);
    } else {
      int kk = (step - 4) * 32;
      tile_load(As, Ww + kk, CW, 64, 64, 32, tid);
      tile_load(Bs, hxb + kk, 64, 64, ncols, 32, tid);
    }
    tile_wait(tid);
    __syncthreads();
    v16bf af = load_afrag(As, mt, lane);
#pragma unroll
    for (int s = 0; s < 2; ++s) {
      v16bf bf = load_bfrag(Bs, LDA, ntb + s, lane);
      acc[s] = wmma_bf16(af, bf, acc[s]);
    }
    __syncthreads();
  }
  int o0 = mt * 16 + 8 * (lane >> 4);
  for (int s = 0; s < 2; ++s) {
    int n = n0 + (ntb + s) * 16 + (lane & 15);
    if (n < XP) {
      Pk8 u;
#pragma unroll
      for (int r = 0; r < 8; ++r) {
        float t = acc[s][r] + WBS[blk * CW + o0 + r];
        if (!last) t = geluf(t);
        u.h8[r] = (__bf16)t;
      }
      *(uint4*)(hXn + ((long)b * XP + n) * 64 + o0) = u.q;
    }
  }
}

// fused head: mid = gelu(fc1 @ h + b1) in LDS (fragment-major), then fc2 scalar
__global__ __launch_bounds__(256) void k_final(const __bf16* hX, const __bf16* FC1B,
    const float* fc1_b, const float* fc2_w, const float* fc2_b, float* out) {
  __shared__ __bf16 As[128 * LDA];
  __shared__ __bf16 Bs[64 * LDA];
  __shared__ __bf16 Mid[64 * MIDS];  // [xl][j]
  int n0 = blockIdx.x * 64, b = blockIdx.y;
  int tid = threadIdx.x, w = tid >> 5, lane = tid & 31;
  const __bf16* hxb = hX + ((long)b * XP + n0) * 64;
  v8f acc[4];
  for (int j = 0; j < 4; ++j) acc[j] = zero8();
  for (int kk = 0; kk < CW; kk += 32) {
    tile_load(As, FC1B + kk, CW, 128, 128, 32, tid);
    tile_load(Bs, hxb + kk, 64, 64, 64, 32, tid);
    tile_wait(tid);
    __syncthreads();
    v16bf af = load_afrag(As, w, lane);
#pragma unroll
    for (int nt = 0; nt < 4; ++nt) {
      v16bf bf = load_bfrag(Bs, LDA, nt, lane);
      acc[nt] = wmma_bf16(af, bf, acc[nt]);
    }
    __syncthreads();
  }
  int o0 = w * 16 + 8 * (lane >> 4);
  for (int nt = 0; nt < 4; ++nt) {
    int n = nt * 16 + (lane & 15);
    Pk8 u;
#pragma unroll
    for (int r = 0; r < 8; ++r)
      u.h8[r] = (__bf16)geluf(acc[nt][r] + fc1_b[o0 + r]);
    *(uint4*)(Mid + n * MIDS + o0) = u.q;
  }
  __syncthreads();
  if (tid < 192) {
    int nc = tid >> 6, xl = tid & 63;
    float a = fc2_b[nc];
    const float* wr = fc2_w + nc * 128;
    const __bf16* mr = Mid + xl * MIDS;
    for (int j = 0; j < 128; ++j) a += wr[j] * (float)mr[j];
    out[((long)b * XF + n0 + xl) * 3 + nc] = a;
  }
}

// ---------------------------------------------------------------- host ------
extern "C" void kernel_launch(void* const* d_in, const int* in_sizes, int n_in,
                              void* d_out, int out_size, void* d_ws, size_t ws_size,
                              hipStream_t stream) {
  const float* x_in   = (const float*)d_in[0];
  const float* grid   = (const float*)d_in[1];
  const float* fc0_w  = (const float*)d_in[2];
  const float* fc0_b  = (const float*)d_in[3];
  const float* fc1_w  = (const float*)d_in[4];
  const float* fc1_b  = (const float*)d_in[5];
  const float* fc2_w  = (const float*)d_in[6];
  const float* fc2_b  = (const float*)d_in[7];
  const float* g_w1   = (const float*)d_in[8];
  const float* g_b1   = (const float*)d_in[9];
  const float* g_w2   = (const float*)d_in[10];
  const float* g_b2   = (const float*)d_in[11];
  const float* w_w    = (const float*)d_in[12];
  const float* w_b    = (const float*)d_in[13];
  const float* log_dc = (const float*)d_in[14];
  const float* mix_re = (const float*)d_in[15];
  const float* mix_im = (const float*)d_in[16];
  const float* aw_raw = (const float*)d_in[17];
  const float* ag_raw = (const float*)d_in[18];
  float* out = (float*)d_out;

  char* p = (char*)d_ws;
  auto carve = [&](size_t bytes) -> void* {
    void* r = (void*)p;
    p += (bytes + 255) & ~(size_t)255;
    return r;
  };
  const size_t HXEL = (size_t)BB * XP * CW;  // x-major activations
  __bf16* hXA  = (__bf16*)carve(HXEL * 2);
  __bf16* hXB  = (__bf16*)carve(HXEL * 2);
  __bf16* GXb  = (__bf16*)carve(HXEL * 2);
  __bf16* DFt  = (__bf16*)carve((size_t)128 * XPS * 2);
  __bf16* DIt  = (__bf16*)carve((size_t)XP * 128 * 2);
  float*  VH   = (float*)carve((size_t)1024 * 128 * 4);
  float*  GH   = (float*)carve((size_t)1024 * 128 * 4);
  __bf16* GRE  = (__bf16*)carve((size_t)65536 * 2);
  __bf16* GIM  = (__bf16*)carve((size_t)65536 * 2);
  float*  GMR  = (float*)carve((size_t)65536 * 4);
  float*  GMI  = (float*)carve((size_t)65536 * 4);
  __bf16* OUTM = (__bf16*)carve((size_t)1024 * 128 * 2);
  __bf16* W1B  = (__bf16*)carve((size_t)NBLK * CH * CW * 2);
  __bf16* W2B  = (__bf16*)carve((size_t)NBLK * CW * CH * 2);
  __bf16* WWB  = (__bf16*)carve((size_t)NBLK * CW * CW * 2);
  __bf16* FC1B = (__bf16*)carve((size_t)CH * CW * 2);
  __bf16* MXRE = (__bf16*)carve((size_t)NBLK * 64 * 64 * 64 * 2);
  __bf16* MXIM = (__bf16*)carve((size_t)NBLK * 64 * 64 * 64 * 2);
  __bf16* MXIMN= (__bf16*)carve((size_t)NBLK * 64 * 64 * 64 * 2);
  float*  EXPZ = (float*)carve((size_t)NBLK * CW * NMODE * 4);
  float*  PF   = (float*)carve((size_t)NBLK * CW * NMODE * 4);
  float*  WBS  = (float*)carve((size_t)NBLK * CW * 4);
  float*  B2S  = (float*)carve((size_t)NBLK * CW * 4);
  float*  SCAL = (float*)carve(8);

  // ---- prep
  k_prep_scalars<<<1, 32, 0, stream>>>(aw_raw, ag_raw, SCAL);
  k_dft_tables<<<(XP + 255) / 256, 256, 0, stream>>>(DFt, DIt);
  k_etd_tables<<<(NBLK * CW * NMODE + 255) / 256, 256, 0, stream>>>(log_dc, EXPZ, PF);
  k_weights<<<(90624 + 255) / 256, 256, 0, stream>>>(g_w1, g_w2, w_w, fc1_w, w_b,
                                                     g_b2, SCAL, W1B, W2B, WWB,
                                                     FC1B, WBS, B2S);
  k_mix<<<(NBLK * 64 * 64 * 64 + 255) / 256, 256, 0, stream>>>(mix_re, mix_im,
                                                               MXRE, MXIM, MXIMN);
  // ---- lift
  k_fc0<<<(int)(((long)BB * XP + 255) / 256), 256, 0, stream>>>(x_in, grid, fc0_w,
                                                                fc0_b, hXA);
  __bf16* cur = hXA; __bf16* nxt = hXB;
  const int NT = (XP + 63) / 64;  // 257
  for (int i = 0; i < NBLK; ++i) {
    k_gfused<<<dim3(NT, BB), 256, 0, stream>>>(cur, GXb, W1B, g_b1, W2B, B2S, i);
    k_zero<<<(1024 * 128 + 255) / 256, 256, 0, stream>>>(VH, 1024 * 128);
    k_zero<<<(1024 * 128 + 255) / 256, 256, 0, stream>>>(GH, 1024 * 128);
    k_dft<<<dim3(16, 2, 8), 256, 0, stream>>>(cur, DFt, VH);
    k_dft<<<dim3(16, 2, 8), 256, 0, stream>>>(GXb, DFt, GH);
    k_gsplit<<<(65536 + 255) / 256, 256, 0, stream>>>(GH, GRE, GIM);
    k_modemix<<<64, 256, 0, stream>>>(GRE, GIM, MXRE, MXIM, MXIMN, GMR, GMI, i);
    k_combine<<<(65536 + 255) / 256, 256, 0, stream>>>(VH, GMR, GMI, EXPZ, PF,
                                                       OUTM, i);
    k_update<<<dim3(NT, BB), 256, 0, stream>>>(OUTM, DIt, WWB, cur, WBS, nxt, i,
                                               i == NBLK - 1 ? 1 : 0);
    __bf16* t = cur; cur = nxt; nxt = t;
  }
  // ---- head
  k_final<<<dim3(XF / 64, BB), 256, 0, stream>>>(cur, FC1B, fc1_b, fc2_w, fc2_b,
                                                 out);
}
